// RiemannFlowMatchingRec_83038897701235
// MI455X (gfx1250) — compile-verified, matrix operand order checked by reading, not run on hardware
//
#include <hip/hip_runtime.h>
#include <cstdint>
#include <cstddef>

#define B_   128
#define L_   200
#define H_   512
#define NH_  8
#define HD_  64
#define FF_  2048
#define MT_  (B_ * L_)          /* 25600 tokens */
#define KP_  224                /* L padded to multiple of 32 for attention */
#define NEGV (-1.0e9f)
#define SCALE_ 0.125f           /* 1/sqrt(64) */
#define LOG1E4 9.210340371976184f

typedef __bf16 bf16_t;
typedef __attribute__((ext_vector_type(16))) __bf16 v16bf;
typedef __attribute__((ext_vector_type(8)))  float  v8f;
typedef int v4i_t __attribute__((vector_size(16)));

// ---- gfx1250 async global->LDS path (guarded; falls back to sync staging) --
#if defined(__has_builtin)
#if __has_builtin(__builtin_amdgcn_global_load_async_to_lds_b128)
#define USE_ASYNC 1
#endif
#endif
#ifndef USE_ASYNC
#define USE_ASYNC 0
#endif

#if USE_ASYNC
// signature (from hipcc diagnostic): (v4i AS(1)*, v4i AS(3)*, imm, imm)
#define ASYNC_COPY_B128(gsrc, ldst)                                        \
  __builtin_amdgcn_global_load_async_to_lds_b128(                          \
      (__attribute__((address_space(1))) v4i_t*)(gsrc),                    \
      (__attribute__((address_space(3))) v4i_t*)(ldst), 0, 0)
#if __has_builtin(__builtin_amdgcn_s_wait_asynccnt)
#define ASYNC_WAIT(n) __builtin_amdgcn_s_wait_asynccnt(n)
#else
#define ASYNC_WAIT(n) asm volatile("s_wait_asynccnt %0" ::"i"(n) : "memory")
#endif
#else
#define ASYNC_COPY_B128(gsrc, ldst) \
  (*(uint4*)(ldst) = *(const uint4*)(gsrc))
#define ASYNC_WAIT(n)
#endif

// ---------------------------------------------------------------- utilities
static __device__ __forceinline__ float wred32(float v) {
#pragma unroll
  for (int o = 16; o > 0; o >>= 1) v += __shfl_xor(v, o, 32);
  return v;
}

// 16-bit WMMA fragment (ISA 7.12.2): per lane, dwords 0..3 = 16 contiguous
// bytes at K = kh*8, dwords 4..7 = 16 contiguous bytes at K = 16 + kh*8.
union FragU { v16bf v; uint4 q[2]; };

static __device__ __forceinline__ void load_frag(FragU& f, const bf16_t* p) {
  f.q[0] = *(const uint4*)p;         // K = kh*8 .. kh*8+7
  f.q[1] = *(const uint4*)(p + 16);  // K = 16+kh*8 .. 16+kh*8+7
}

// ---------------------------------------------------------------- weight prep
__global__ __launch_bounds__(256) void transpose_to_bf16_kernel(
    const float* __restrict__ W, bf16_t* __restrict__ Wt, int K, int N) {
  int idx = blockIdx.x * 256 + threadIdx.x;
  if (idx < K * N) {
    int k = idx / N, n = idx % N;
    Wt[(size_t)n * K + k] = (bf16_t)W[idx];
  }
}

// ---------------------------------------------------------------- WMMA GEMM
// C[M,N] = act(A[M,K] @ Wt[N,K]^T + bias[N]) (+ residual)
// block = 256 threads (8 waves), block tile 128x128, wave tile 32x64 (2x4),
// K-step 32, double-buffered LDS staged via async global->LDS.
template <int ACT, bool HAS_RES, bool OUTF, bool OUTB>
__global__ __launch_bounds__(256) void gemm_wmma_kernel(
    const bf16_t* __restrict__ A, const bf16_t* __restrict__ Wt,
    const float* __restrict__ bias, const float* __restrict__ res,
    float* __restrict__ outF, bf16_t* __restrict__ outB,
    int K, int lda, int ldcF, int ldcB, int coloffB) {
  __shared__ bf16_t sA[2][128 * 32];
  __shared__ bf16_t sB[2][128 * 32];
  const int tid  = threadIdx.x;
  const int lane = tid & 31;
  const int wave = tid >> 5;
  const int m0 = blockIdx.y * 128;
  const int n0 = blockIdx.x * 128;
  const int wm = (wave >> 1) * 32;  // 0,32,64,96
  const int wn = (wave & 1) * 64;   // 0,64
  const int fr = lane & 15;
  const int kh = lane >> 4;

  v8f acc[2][4];
#pragma unroll
  for (int i = 0; i < 2; ++i)
#pragma unroll
    for (int j = 0; j < 4; ++j)
#pragma unroll
      for (int e = 0; e < 8; ++e) acc[i][j][e] = 0.0f;

  // stage one 128x32 K-slab of A and Wt into LDS buffer `buf`
  auto stage = [&](int buf, int k0) {
#pragma unroll
    for (int it = 0; it < 2; ++it) {
      int i = tid + it * 256;        // 0..511
      int r = i >> 2;                // 0..127
      int c = (i & 3) << 3;          // 0,8,16,24
      ASYNC_COPY_B128(&A[(size_t)(m0 + r) * lda + k0 + c],
                      &sA[buf][r * 32 + c]);
      ASYNC_COPY_B128(&Wt[(size_t)(n0 + r) * K + k0 + c],
                      &sB[buf][r * 32 + c]);
    }
  };

  const int nk = K >> 5;
  stage(0, 0);
  for (int kt = 0; kt < nk; ++kt) {
    const int buf = kt & 1;
    if (kt + 1 < nk) {
      stage(buf ^ 1, (kt + 1) * 32);
      ASYNC_WAIT(4);  // 4 newest (next slab) may remain in flight
    } else {
      ASYNC_WAIT(0);
    }
    __syncthreads();

    FragU af[2], bf_[4];
#pragma unroll
    for (int f = 0; f < 2; ++f)
      load_frag(af[f], &sA[buf][(wm + f * 16 + fr) * 32 + kh * 8]);
#pragma unroll
    for (int f = 0; f < 4; ++f)
      load_frag(bf_[f], &sB[buf][(wn + f * 16 + fr) * 32 + kh * 8]);
#pragma unroll
    for (int fm = 0; fm < 2; ++fm)
#pragma unroll
      for (int fn = 0; fn < 4; ++fn)
        acc[fm][fn] = __builtin_amdgcn_wmma_f32_16x16x32_bf16(
            false, af[fm].v, false, bf_[fn].v, (short)0, acc[fm][fn],
            false, false);
    __syncthreads();
  }

  // epilogue
#pragma unroll
  for (int fm = 0; fm < 2; ++fm)
#pragma unroll
    for (int fn = 0; fn < 4; ++fn)
#pragma unroll
      for (int r = 0; r < 8; ++r) {
        int m = m0 + wm + fm * 16 + kh * 8 + r;
        int n = n0 + wn + fn * 16 + fr;
        float v = acc[fm][fn][r] + bias[n];
        if (ACT == 1) {  // tanh-approx GELU
          float xx = v;
          v = 0.5f * xx *
              (1.0f + tanhf(0.7978845608028654f * (xx + 0.044715f * xx * xx * xx)));
        } else if (ACT == 2) {  // SiLU
          v = v / (1.0f + __expf(-v));
        }
        if (HAS_RES) v += res[(size_t)m * ldcF + n];
        if (OUTF) outF[(size_t)m * ldcF + n] = v;
        if (OUTB) outB[(size_t)m * ldcB + coloffB + n] = (bf16_t)v;
      }
}

// ---------------------------------------------------------------- LayerNorm
// 1 wave / token, 8 tokens / block; lane handles 16 strided columns
__global__ __launch_bounds__(256) void ln_kernel(
    const float* __restrict__ x, const float* __restrict__ g,
    const float* __restrict__ bta, bf16_t* __restrict__ outB,
    float* __restrict__ outF) {
  const int tok  = blockIdx.x * 8 + (threadIdx.x >> 5);
  const int lane = threadIdx.x & 31;
  const float* row = x + (size_t)tok * H_;
  float vals[16];
  float s = 0.f, ss = 0.f;
#pragma unroll
  for (int j = 0; j < 16; ++j) {
    float v = row[lane + 32 * j];
    vals[j] = v; s += v; ss += v * v;
  }
  s = wred32(s); ss = wred32(ss);
  float mean = s * (1.0f / H_);
  float var  = ss * (1.0f / H_) - mean * mean;
  float rstd = rsqrtf(var + 1e-5f);
#pragma unroll
  for (int j = 0; j < 16; ++j) {
    int col = lane + 32 * j;
    float o = (vals[j] - mean) * rstd * g[col] + bta[col];
    if (outB) outB[(size_t)tok * H_ + col] = (bf16_t)o;
    if (outF) outF[(size_t)tok * H_ + col] = o;
  }
}

// LN + flow-matching per-token loss:  fm = ||LN(pre) - v_t||^2 / H
__global__ __launch_bounds__(256) void ln_loss_kernel(
    const float* __restrict__ pre, const float* __restrict__ g,
    const float* __restrict__ bta, const float* __restrict__ vt,
    float* __restrict__ fm) {
  const int tok  = blockIdx.x * 8 + (threadIdx.x >> 5);
  const int lane = threadIdx.x & 31;
  const float* row = pre + (size_t)tok * H_;
  float vals[16];
  float s = 0.f, ss = 0.f;
#pragma unroll
  for (int j = 0; j < 16; ++j) {
    float v = row[lane + 32 * j];
    vals[j] = v; s += v; ss += v * v;
  }
  s = wred32(s); ss = wred32(ss);
  float mean = s * (1.0f / H_);
  float rstd = rsqrtf(ss * (1.0f / H_) - mean * mean + 1e-5f);
  float d2 = 0.f;
#pragma unroll
  for (int j = 0; j < 16; ++j) {
    int col = lane + 32 * j;
    float p = (vals[j] - mean) * rstd * g[col] + bta[col];
    float d = p - vt[(size_t)tok * H_ + col];
    d2 += d * d;
  }
  d2 = wred32(d2);
  if (lane == 0) fm[tok] = d2 * (1.0f / H_);
}

// ---------------------------------------------------------------- RoPE (+bf16)
__global__ __launch_bounds__(256) void rope_kernel(
    const bf16_t* __restrict__ qr, const bf16_t* __restrict__ kr,
    bf16_t* __restrict__ qo, bf16_t* __restrict__ ko) {
  const int tok  = blockIdx.x;
  const int head = threadIdx.x >> 5;
  const int j    = threadIdx.x & 31;
  const int l    = tok % L_;
  float inv = __expf(-LOG1E4 * (float)(2 * j) / (float)HD_);
  float fr  = (float)l * inv;
  float c = cosf(fr), sn = sinf(fr);
  size_t base = (size_t)tok * H_ + head * HD_;
  float qe = (float)qr[base + 2 * j], qoo = (float)qr[base + 2 * j + 1];
  qo[base + j]      = (bf16_t)(qe * c - qoo * sn);
  qo[base + 32 + j] = (bf16_t)(qe * sn + qoo * c);
  float ke = (float)kr[base + 2 * j], koo = (float)kr[base + 2 * j + 1];
  ko[base + j]      = (bf16_t)(ke * c - koo * sn);
  ko[base + 32 + j] = (bf16_t)(ke * sn + koo * c);
}

// ---------------------------------------------------------------- attention
// one wave per (batch, head, 16-query tile); WMMA scores + softmax + WMMA a@V
__global__ __launch_bounds__(32) void attn_kernel(
    const bf16_t* __restrict__ qb, const bf16_t* __restrict__ kb,
    const bf16_t* __restrict__ vb, const int* __restrict__ msk,
    bf16_t* __restrict__ ob) {
  __shared__ bf16_t vt[HD_ * KP_];   // V transposed [d][key]
  __shared__ float  sc[16 * KP_];    // score panel
  __shared__ bf16_t aw[16 * KP_];    // softmax weights bf16
  const int lane = threadIdx.x;
  const int q0   = blockIdx.x * 16;
  const int head = blockIdx.y;
  const int b    = blockIdx.z;
  const int fr = lane & 15;
  const int kh = lane >> 4;
  const size_t baseRow = (size_t)b * L_ * H_ + head * HD_;

  // stage V^T
  for (int i = lane; i < HD_ * KP_; i += 32) {
    int d = i / KP_, key = i % KP_;
    bf16_t val = (bf16_t)0.0f;
    if (key < L_) val = vb[baseRow + (size_t)key * H_ + d];
    vt[d * KP_ + key] = val;
  }

  // Q fragments, two 32-wide chunks over d (each = two b128 loads)
  FragU qf[2];
  const int qrow = q0 + fr;
  uint4 z4; z4.x = z4.y = z4.z = z4.w = 0u;
#pragma unroll
  for (int c = 0; c < 2; ++c) {
    if (qrow < L_)
      load_frag(qf[c], &qb[baseRow + (size_t)qrow * H_ + c * 32 + kh * 8]);
    else { qf[c].q[0] = z4; qf[c].q[1] = z4; }
  }
  __syncthreads();

  // scores: loop key tiles of 16
  for (int kt = 0; kt < KP_ / 16; ++kt) {
    const int key = kt * 16 + fr;
    FragU kf[2];
#pragma unroll
    for (int c = 0; c < 2; ++c) {
      if (key < L_)
        load_frag(kf[c], &kb[baseRow + (size_t)key * H_ + c * 32 + kh * 8]);
      else { kf[c].q[0] = z4; kf[c].q[1] = z4; }
    }
    v8f a8;
#pragma unroll
    for (int e = 0; e < 8; ++e) a8[e] = 0.0f;
    a8 = __builtin_amdgcn_wmma_f32_16x16x32_bf16(false, qf[0].v, false, kf[0].v,
                                                 (short)0, a8, false, false);
    a8 = __builtin_amdgcn_wmma_f32_16x16x32_bf16(false, qf[1].v, false, kf[1].v,
                                                 (short)0, a8, false, false);
    int mv = (key < L_) ? msk[b * L_ + key] : 0;
#pragma unroll
    for (int r = 0; r < 8; ++r) {
      int mrow = kh * 8 + r;
      int qg = q0 + mrow;
      bool dead = (key >= L_) || (key > qg) || (mv == 0);
      sc[mrow * KP_ + key] = dead ? NEGV : a8[r] * SCALE_;
    }
  }
  __syncthreads();

  // softmax, lanes 0..15 each own a query row
  if (lane < 16) {
    float mx = -3.0e38f;
    for (int k = 0; k < KP_; ++k) mx = fmaxf(mx, sc[lane * KP_ + k]);
    float sum = 0.f;
    for (int k = 0; k < KP_; ++k) {
      float e = __expf(sc[lane * KP_ + k] - mx);
      sc[lane * KP_ + k] = e;
      sum += e;
    }
    float inv = 1.0f / sum;
    for (int k = 0; k < KP_; ++k)
      aw[lane * KP_ + k] = (bf16_t)(sc[lane * KP_ + k] * inv);
  }
  __syncthreads();

  // O = A @ V, d tiles of 16, K chunks of 32 keys
  for (int dt = 0; dt < HD_ / 16; ++dt) {
    v8f o8;
#pragma unroll
    for (int e = 0; e < 8; ++e) o8[e] = 0.0f;
    for (int kc = 0; kc < KP_ / 32; ++kc) {
      FragU af, bf_;
      load_frag(af, &aw[fr * KP_ + kc * 32 + kh * 8]);
      load_frag(bf_, &vt[(dt * 16 + fr) * KP_ + kc * 32 + kh * 8]);
      o8 = __builtin_amdgcn_wmma_f32_16x16x32_bf16(false, af.v, false, bf_.v,
                                                   (short)0, o8, false, false);
    }
#pragma unroll
    for (int r = 0; r < 8; ++r) {
      int qg = q0 + kh * 8 + r;
      if (qg < L_)
        ob[baseRow + (size_t)qg * H_ + dt * 16 + fr] = (bf16_t)o8[r];
    }
  }
}

// ---------------------------------------------------------------- flow head
__global__ __launch_bounds__(256) void flow_kernel(
    const float* __restrict__ tgt, const float* __restrict__ seq,
    const float* __restrict__ tv, bf16_t* __restrict__ hin,
    float* __restrict__ vt) {
  const int tok  = blockIdx.x * 8 + (threadIdx.x >> 5);
  const int lane = threadIdx.x & 31;
  const float* z0r = tgt + (size_t)tok * H_;
  const float* z1r = seq + (size_t)tok * H_;
  float a0[16], a1[16];
  float s0 = 0.f, s1 = 0.f;
#pragma unroll
  for (int j = 0; j < 16; ++j) {
    int col = lane + 32 * j;
    a0[j] = z0r[col]; a1[j] = z1r[col];
    s0 += a0[j] * a0[j]; s1 += a1[j] * a1[j];
  }
  s0 = wred32(s0); s1 = wred32(s1);
  float in0 = 1.0f / fmaxf(sqrtf(s0), 1e-12f);
  float in1 = 1.0f / fmaxf(sqrtf(s1), 1e-12f);
  float inner = 0.f;
#pragma unroll
  for (int j = 0; j < 16; ++j) {
    a0[j] *= in0; a1[j] *= in1;
    inner += a0[j] * a1[j];
  }
  inner = wred32(inner);
  inner = fminf(fmaxf(inner, -1.0f + 1e-6f), 1.0f - 1e-6f);
  float dg = acosf(inner);
  float w[16];
  float dotxw = 0.f;
#pragma unroll
  for (int j = 0; j < 16; ++j) { w[j] = a1[j] - a0[j]; dotxw += a0[j] * w[j]; }
  dotxw = wred32(dotxw);
  float nu2 = 0.f;
#pragma unroll
  for (int j = 0; j < 16; ++j) { w[j] -= dotxw * a0[j]; nu2 += w[j] * w[j]; }
  nu2 = wred32(nu2);
  float nu = fmaxf(sqrtf(nu2), 1e-6f);
  float su = (dg > 1e-6f) ? dg / nu : 1.0f;
  float un2 = 0.f;
#pragma unroll
  for (int j = 0; j < 16; ++j) { w[j] *= su; un2 += w[j] * w[j]; }
  un2 = wred32(un2);
  float n = sqrtf(un2);
  float tt = tv[tok];
  float s = tt * n;
  float zc, usc, vz0, vu;
  if (s > 1e-6f) {
    float cs = cosf(s), sn = sinf(s);
    zc = cs; usc = sn / n;     // z_t = z0*cos(s) + u*sin(s)/n
    vz0 = -n * sn; vu = cs;    // v_t = -z0*n*sin(s) + u*cos(s)
  } else {
    zc = 1.0f; usc = tt; vz0 = 0.0f; vu = 1.0f;
  }
#pragma unroll
  for (int j = 0; j < 16; ++j) {
    int col = lane + 32 * j;
    hin[(size_t)tok * (2 * H_) + col] = (bf16_t)(a0[j] * zc + w[j] * usc);
    vt[(size_t)tok * H_ + col] = a0[j] * vz0 + w[j] * vu;
  }
}

// sinusoidal time embedding -> bf16 [MT, 512]
__global__ __launch_bounds__(256) void time_emb_kernel(
    const float* __restrict__ tv, bf16_t* __restrict__ emb) {
  const int tok = blockIdx.x;
  const int j   = threadIdx.x;  // 0..255
  float ts = tv[tok] * 1000.0f;
  float fq = __expf(-LOG1E4 * (float)j / 256.0f);
  float a = ts * fq;
  emb[(size_t)tok * H_ + j]       = (bf16_t)cosf(a);
  emb[(size_t)tok * H_ + 256 + j] = (bf16_t)sinf(a);
}

// ---------------------------------------------------------------- loss reduce
__global__ __launch_bounds__(256) void batch_reduce_kernel(
    const float* __restrict__ fm, const float* __restrict__ mask,
    float* __restrict__ partial) {
  __shared__ float sm[256], sa[256];
  const int b = blockIdx.x, tid = threadIdx.x;
  float m = 0.f, a = 0.f;
  for (int l = tid; l < L_; l += 256) {
    float wv = mask[b * L_ + l];
    m += wv; a += wv * fm[b * L_ + l];
  }
  sm[tid] = m; sa[tid] = a;
  __syncthreads();
  for (int off = 128; off > 0; off >>= 1) {
    if (tid < off) { sm[tid] += sm[tid + off]; sa[tid] += sa[tid + off]; }
    __syncthreads();
  }
  if (tid == 0) partial[b] = sa[0] / sm[0];
}

__global__ __launch_bounds__(128) void final_reduce_kernel(
    const float* __restrict__ partial, float* __restrict__ out) {
  __shared__ float s[128];
  const int tid = threadIdx.x;
  s[tid] = partial[tid];
  __syncthreads();
  for (int off = 64; off > 0; off >>= 1) {
    if (tid < off) s[tid] += s[tid + off];
    __syncthreads();
  }
  if (tid == 0) out[0] = s[0] / (float)B_;
}

// ---------------------------------------------------------------- host side
static void run_gemm(hipStream_t st, int act, const bf16_t* A, const bf16_t* Wt,
                     const float* bias, const float* res, float* outF,
                     bf16_t* outB, int M, int N, int K, int lda, int ldcF,
                     int ldcB, int coloffB) {
  dim3 grid(N / 128, M / 128), blk(256);
  if (res)
    gemm_wmma_kernel<0, true, true, false><<<grid, blk, 0, st>>>(
        A, Wt, bias, res, outF, outB, K, lda, ldcF, ldcB, coloffB);
  else if (act == 1)
    gemm_wmma_kernel<1, false, false, true><<<grid, blk, 0, st>>>(
        A, Wt, bias, res, outF, outB, K, lda, ldcF, ldcB, coloffB);
  else if (act == 2)
    gemm_wmma_kernel<2, false, false, true><<<grid, blk, 0, st>>>(
        A, Wt, bias, res, outF, outB, K, lda, ldcF, ldcB, coloffB);
  else if (outB)
    gemm_wmma_kernel<0, false, false, true><<<grid, blk, 0, st>>>(
        A, Wt, bias, res, outF, outB, K, lda, ldcF, ldcB, coloffB);
  else
    gemm_wmma_kernel<0, false, true, false><<<grid, blk, 0, st>>>(
        A, Wt, bias, res, outF, outB, K, lda, ldcF, ldcB, coloffB);
}

extern "C" void kernel_launch(void* const* d_in, const int* in_sizes, int n_in,
                              void* d_out, int out_size, void* d_ws,
                              size_t ws_size, hipStream_t stream) {
  (void)in_sizes; (void)n_in; (void)out_size; (void)ws_size;
  const float* input_embs  = (const float*)d_in[0];
  const float* target_embs = (const float*)d_in[1];
  const int*   input_mask  = (const int*)d_in[2];
  const float* target_mask = (const float*)d_in[3];
  const float* tvals       = (const float*)d_in[4];
  const float* Wq = (const float*)d_in[5];
  const float* Wk = (const float*)d_in[6];
  const float* Wv = (const float*)d_in[7];
  const float* Wo = (const float*)d_in[8];
  const float* bq = (const float*)d_in[9];
  const float* bk = (const float*)d_in[10];
  const float* bv = (const float*)d_in[11];
  const float* bo = (const float*)d_in[12];
  const float* W1 = (const float*)d_in[13];
  const float* b1 = (const float*)d_in[14];
  const float* W2 = (const float*)d_in[15];
  const float* b2 = (const float*)d_in[16];
  const float* ln1_g = (const float*)d_in[17];
  const float* ln1_b = (const float*)d_in[18];
  const float* ln2_g = (const float*)d_in[19];
  const float* ln2_b = (const float*)d_in[20];
  const float* lnf_g = (const float*)d_in[21];
  const float* lnf_b = (const float*)d_in[22];
  const float* dln_g = (const float*)d_in[23];
  const float* dln_b = (const float*)d_in[24];
  const float* tW1 = (const float*)d_in[25];
  const float* tb1 = (const float*)d_in[26];
  const float* tW2 = (const float*)d_in[27];
  const float* tb2 = (const float*)d_in[28];
  const float* dW1 = (const float*)d_in[29];
  const float* db1 = (const float*)d_in[30];
  const float* dW2 = (const float*)d_in[31];
  const float* db2 = (const float*)d_in[32];

  float* out = (float*)d_out;
  float* seq = out + 1;  // d_out[0] = loss, d_out[1..] = seq (B*L*H f32)

  // workspace bump allocator (256B aligned)
  char* base = (char*)d_ws;
  size_t off = 0;
  auto alloc = [&](size_t bytes) -> char* {
    char* p = base + off;
    off += (bytes + 255) & ~(size_t)255;
    return p;
  };
  const size_t SW  = (size_t)H_ * H_;        // 512*512
  const size_t SF1 = (size_t)H_ * FF_;       // 512*2048
  bf16_t* WqT  = (bf16_t*)alloc(2 * SW * 2);
  bf16_t* WkT  = (bf16_t*)alloc(2 * SW * 2);
  bf16_t* WvT  = (bf16_t*)alloc(2 * SW * 2);
  bf16_t* WoT  = (bf16_t*)alloc(2 * SW * 2);
  bf16_t* W1T  = (bf16_t*)alloc(2 * SF1 * 2);
  bf16_t* W2T  = (bf16_t*)alloc(2 * SF1 * 2);
  bf16_t* tW1T = (bf16_t*)alloc(SF1 * 2);
  bf16_t* tW2T = (bf16_t*)alloc(SF1 * 2);
  bf16_t* dW1T = (bf16_t*)alloc(2 * SF1 * 2);
  bf16_t* dW2T = (bf16_t*)alloc(SF1 * 2);
  float*  x    = (float*)alloc((size_t)MT_ * H_ * 4);
  bf16_t* h    = (bf16_t*)alloc((size_t)MT_ * H_ * 2);
  bf16_t* qraw = (bf16_t*)alloc((size_t)MT_ * H_ * 2);
  bf16_t* kraw = (bf16_t*)alloc((size_t)MT_ * H_ * 2);
  bf16_t* qb   = (bf16_t*)alloc((size_t)MT_ * H_ * 2);
  bf16_t* kb   = (bf16_t*)alloc((size_t)MT_ * H_ * 2);
  bf16_t* vb   = (bf16_t*)alloc((size_t)MT_ * H_ * 2);
  bf16_t* obf  = (bf16_t*)alloc((size_t)MT_ * H_ * 2);
  bf16_t* big  = (bf16_t*)alloc((size_t)MT_ * FF_ * 2);
  bf16_t* emb  = (bf16_t*)alloc((size_t)MT_ * H_ * 2);
  bf16_t* hin  = (bf16_t*)alloc((size_t)MT_ * 2 * H_ * 2);
  float*  vtb  = (float*)alloc((size_t)MT_ * H_ * 4);
  float*  pre  = (float*)alloc((size_t)MT_ * H_ * 4);
  float*  fm   = (float*)alloc((size_t)MT_ * 4);
  float*  part = (float*)alloc((size_t)B_ * 4);

  // weight prep: fp32 [K,N] -> bf16 [N,K]
  auto tr = [&](const float* W, bf16_t* Wt, int K, int N) {
    int n = K * N;
    transpose_to_bf16_kernel<<<(n + 255) / 256, 256, 0, stream>>>(W, Wt, K, N);
  };
  for (int i = 0; i < 2; ++i) {
    tr(Wq + i * SW, WqT + i * SW, H_, H_);
    tr(Wk + i * SW, WkT + i * SW, H_, H_);
    tr(Wv + i * SW, WvT + i * SW, H_, H_);
    tr(Wo + i * SW, WoT + i * SW, H_, H_);
    tr(W1 + i * SF1, W1T + i * SF1, H_, FF_);
    tr(W2 + i * SF1, W2T + i * SF1, FF_, H_);
  }
  tr(tW1, tW1T, H_, FF_);
  tr(tW2, tW2T, FF_, H_);
  tr(dW1, dW1T, 2 * H_, FF_);
  tr(dW2, dW2T, FF_, H_);

  // residual stream init
  (void)hipMemcpyAsync(x, input_embs, (size_t)MT_ * H_ * sizeof(float),
                       hipMemcpyDeviceToDevice, stream);

  const int LNB = MT_ / 8;
  for (int i = 0; i < 2; ++i) {
    ln_kernel<<<LNB, 256, 0, stream>>>(x, ln1_g + i * H_, ln1_b + i * H_, h,
                                       nullptr);
    run_gemm(stream, 0, h, WqT + i * SW, bq + i * H_, nullptr, nullptr, qraw,
             MT_, H_, H_, H_, H_, H_, 0);
    run_gemm(stream, 0, h, WkT + i * SW, bk + i * H_, nullptr, nullptr, kraw,
             MT_, H_, H_, H_, H_, H_, 0);
    run_gemm(stream, 0, h, WvT + i * SW, bv + i * H_, nullptr, nullptr, vb,
             MT_, H_, H_, H_, H_, H_, 0);
    rope_kernel<<<MT_, 256, 0, stream>>>(qraw, kraw, qb, kb);
    attn_kernel<<<dim3((L_ + 15) / 16, NH_, B_), 32, 0, stream>>>(
        qb, kb, vb, input_mask, obf);
    run_gemm(stream, 0, obf, WoT + i * SW, bo + i * H_, x, x, nullptr, MT_, H_,
             H_, H_, H_, 0, 0);
    ln_kernel<<<LNB, 256, 0, stream>>>(x, ln2_g + i * H_, ln2_b + i * H_, h,
                                       nullptr);
    run_gemm(stream, 1, h, W1T + i * SF1, b1 + i * FF_, nullptr, nullptr, big,
             MT_, FF_, H_, H_, H_, FF_, 0);
    run_gemm(stream, 0, big, W2T + i * SF1, b2 + i * H_, x, x, nullptr, MT_,
             H_, FF_, FF_, H_, 0, 0);
  }
  // final LN -> seq (f32 output straight into d_out[1..])
  ln_kernel<<<LNB, 256, 0, stream>>>(x, lnf_g, lnf_b, nullptr, seq);

  // flow head: z_t (bf16 -> hin[:, :512]) and v_t (f32)
  flow_kernel<<<LNB, 256, 0, stream>>>(target_embs, seq, tvals, hin, vtb);
  time_emb_kernel<<<MT_, 256, 0, stream>>>(tvals, emb);
  run_gemm(stream, 2, emb, tW1T, tb1, nullptr, nullptr, big, MT_, FF_, H_, H_,
           H_, FF_, 0);
  run_gemm(stream, 0, big, tW2T, tb2, nullptr, nullptr, hin, MT_, H_, FF_, FF_,
           H_, 2 * H_, H_);  // time_emb -> hin[:, 512:1024]
  run_gemm(stream, 2, hin, dW1T, db1, nullptr, nullptr, big, MT_, FF_, 2 * H_,
           2 * H_, H_, FF_, 0);
  run_gemm(stream, 0, big, dW2T, db2, nullptr, pre, nullptr, MT_, H_, FF_, FF_,
           H_, 0, 0);
  ln_loss_kernel<<<LNB, 256, 0, stream>>>(pre, dln_g, dln_b, vtb, fm);
  batch_reduce_kernel<<<B_, 256, 0, stream>>>(fm, target_mask, part);
  final_reduce_kernel<<<1, 128, 0, stream>>>(part, out);
}
